// LongcatRouter_49065706390112
// MI455X (gfx1250) — compile-verified, matrix operand order checked by reading, not run on hardware
//
#include <hip/hip_runtime.h>
#include <hip/hip_bf16.h>
#include <stdint.h>

typedef __bf16 bf16_t;
typedef __attribute__((ext_vector_type(8)))  __bf16 bf16x8;
typedef __attribute__((ext_vector_type(16))) __bf16 bf16x16;
typedef __attribute__((ext_vector_type(8)))  float  f32x8;

#define HIDDEN_DIM 6144
#define BM 128
#define BN 128
#define BK 32
#define PAD 8
#define LDSS (BK + PAD)           // 40 bf16 row stride: slot = 5r mod 16 is a permutation
#define NKT (HIDDEN_DIM / BK)     // 192 K-tiles

// ---- CDNA5 async global->LDS path (ASYNCcnt); probe-confirmed on this toolchain.
#if defined(__has_builtin)
#if __has_builtin(__builtin_amdgcn_global_load_async_to_lds_b128)
#define USE_ASYNC_LDS 1
#endif
#endif

// Builtin signature (from hipcc diagnostic): (int4 AS1*, int4 AS3*, imm int, imm int)
typedef int v4i __attribute__((vector_size(16)));
typedef __attribute__((address_space(1))) v4i* gv4i_p;
typedef __attribute__((address_space(3))) v4i* sv4i_p;

__device__ __forceinline__ void async_wait0() {
#if defined(__has_builtin) && __has_builtin(__builtin_amdgcn_s_wait_asynccnt)
    __builtin_amdgcn_s_wait_asynccnt(0);
#else
    asm volatile("s_wait_asynccnt 0x0" ::: "memory");
#endif
}

// Stage one BK=32 K-slab of the A (128 token rows) and B/W (128 expert rows)
// tiles into LDS. gA/gB are already offset to (tileRow + ldRow)*HIDDEN + ldCol.
// 256 threads x 16B x 2 passes cover each 128x32 bf16 (8 KB) tile.
__device__ __forceinline__ void stage_tiles(bf16_t* sAbuf, bf16_t* sBbuf,
                                            const bf16_t* gA, const bf16_t* gB,
                                            int ldRow, int ldCol, int kOff)
{
#ifdef USE_ASYNC_LDS
#pragma unroll
    for (int i = 0; i < 2; ++i)
        __builtin_amdgcn_global_load_async_to_lds_b128(
            (gv4i_p)(uintptr_t)(gA + (size_t)(i * 64) * HIDDEN_DIM + kOff),
            (sv4i_p)(uint32_t)(uintptr_t)&sAbuf[(ldRow + i * 64) * LDSS + ldCol],
            0, 0);
#pragma unroll
    for (int i = 0; i < 2; ++i)
        __builtin_amdgcn_global_load_async_to_lds_b128(
            (gv4i_p)(uintptr_t)(gB + (size_t)(i * 64) * HIDDEN_DIM + kOff),
            (sv4i_p)(uint32_t)(uintptr_t)&sBbuf[(ldRow + i * 64) * LDSS + ldCol],
            0, 0);
#else
#pragma unroll
    for (int i = 0; i < 2; ++i) {
        uint4 t = *reinterpret_cast<const uint4*>(gA + (size_t)(i * 64) * HIDDEN_DIM + kOff);
        *reinterpret_cast<uint4*>(&sAbuf[(ldRow + i * 64) * LDSS + ldCol]) = t;
    }
#pragma unroll
    for (int i = 0; i < 2; ++i) {
        uint4 t = *reinterpret_cast<const uint4*>(gB + (size_t)(i * 64) * HIDDEN_DIM + kOff);
        *reinterpret_cast<uint4*>(&sBbuf[(ldRow + i * 64) * LDSS + ldCol]) = t;
    }
#endif
}

__global__ __launch_bounds__(256, 2)
void router_gemm_bf16_wmma(const bf16_t* __restrict__ A,
                           const bf16_t* __restrict__ W,
                           float* __restrict__ C,
                           int E)
{
    // Double-buffered tiles: 2*2*128*40*2B = 40,960 B of LDS.
    __shared__ __align__(16) bf16_t sA[2][BM * LDSS];
    __shared__ __align__(16) bf16_t sB[2][BN * LDSS];

    const int tid  = threadIdx.x;
    const int lane = tid & 31;
    const int wave = tid >> 5;
    const int wm   = wave & 3;     // 4 waves along M: 32 rows each
    const int wn   = wave >> 2;    // 2 waves along N: 64 cols each

    const int r  = lane & 15;      // fragment row (A) / column (B)
    const int h  = lane >> 4;      // lane half
    const int kb = h * 8;          // K sub-offset per ISA 16-bit operand layout

    const int tileN = blockIdx.x * BN;
    const int tileM = blockIdx.y * BM;

    // Copy mapping: thread t -> row t/4, 16B chunk (t%4); 64 rows per pass.
    const int ldRow = tid >> 2;          // 0..63
    const int ldCol = (tid & 3) * 8;     // 0,8,16,24 (bf16 elements)

    const bf16_t* gA = A + (size_t)(tileM + ldRow) * HIDDEN_DIM + ldCol;
    const bf16_t* gB = W + (size_t)(tileN + ldRow) * HIDDEN_DIM + ldCol;

    f32x8 acc[2][4];
#pragma unroll
    for (int i = 0; i < 2; ++i)
#pragma unroll
        for (int j = 0; j < 4; ++j)
#pragma unroll
            for (int v = 0; v < 8; ++v) acc[i][j][v] = 0.0f;

    // Prologue: kick off copy of K-tile 0 into buffer 0.
    stage_tiles(sA[0], sB[0], gA, gB, ldRow, ldCol, 0);

    for (int kt = 0; kt < NKT; ++kt) {
        const int cur = kt & 1;

#ifdef USE_ASYNC_LDS
        async_wait0();               // my portion of tile kt has landed in LDS
#endif
        __syncthreads();             // everyone's portion landed; all reads of the
                                     // other buffer (iter kt-1) are complete

        if (kt + 1 < NKT)            // overlap next copy with this tile's math
            stage_tiles(sA[cur ^ 1], sB[cur ^ 1], gA, gB, ldRow, ldCol, (kt + 1) * BK);

        const bf16_t* bA = sA[cur];
        const bf16_t* bB = sB[cur];

        bf16x16 af[2];
        bf16x16 bfrag[4];
#pragma unroll
        for (int mi = 0; mi < 2; ++mi) {
            const int row = wm * 32 + mi * 16 + r;
            bf16x8 lo = *reinterpret_cast<const bf16x8*>(&bA[row * LDSS + kb]);
            bf16x8 hi = *reinterpret_cast<const bf16x8*>(&bA[row * LDSS + kb + 16]);
            af[mi] = __builtin_shufflevector(lo, hi,
                     0,1,2,3,4,5,6,7,8,9,10,11,12,13,14,15);
        }
#pragma unroll
        for (int ni = 0; ni < 4; ++ni) {
            const int row = wn * 64 + ni * 16 + r;   // expert row of W == B column
            bf16x8 lo = *reinterpret_cast<const bf16x8*>(&bB[row * LDSS + kb]);
            bf16x8 hi = *reinterpret_cast<const bf16x8*>(&bB[row * LDSS + kb + 16]);
            bfrag[ni] = __builtin_shufflevector(lo, hi,
                        0,1,2,3,4,5,6,7,8,9,10,11,12,13,14,15);
        }
#pragma unroll
        for (int mi = 0; mi < 2; ++mi)
#pragma unroll
            for (int ni = 0; ni < 4; ++ni)
                acc[mi][ni] = __builtin_amdgcn_wmma_f32_16x16x32_bf16(
                    false, af[mi], false, bfrag[ni],
                    (short)0, acc[mi][ni], false, false);
    }

    // Epilogue: C layout — lane half h -> M base (0/8), VGPR v -> M offset, r -> N.
    const int colBase = tileN + wn * 64 + r;
    const int rowBase = tileM + wm * 32 + h * 8;
#pragma unroll
    for (int mi = 0; mi < 2; ++mi)
#pragma unroll
        for (int ni = 0; ni < 4; ++ni) {
            const int col = colBase + ni * 16;
#pragma unroll
            for (int v = 0; v < 8; ++v) {
                const int row = rowBase + mi * 16 + v;
                C[(size_t)row * E + col] = acc[mi][ni][v];
            }
        }
}

extern "C" void kernel_launch(void* const* d_in, const int* in_sizes, int n_in,
                              void* d_out, int out_size, void* d_ws, size_t ws_size,
                              hipStream_t stream) {
    (void)n_in; (void)out_size; (void)d_ws; (void)ws_size;
    const bf16_t* A = reinterpret_cast<const bf16_t*>(d_in[0]);
    const bf16_t* W = reinterpret_cast<const bf16_t*>(d_in[1]);
    float* C = reinterpret_cast<float*>(d_out);
    const int T = in_sizes[0] / HIDDEN_DIM;   // 16384 tokens
    const int E = in_sizes[1] / HIDDEN_DIM;   // 768 experts
    dim3 grid((unsigned)(E / BN), (unsigned)(T / BM));
    router_gemm_bf16_wmma<<<grid, dim3(256), 0, stream>>>(A, W, C, E);
}
